// LinearRTU_Triton_84920093377115
// MI455X (gfx1250) — compile-verified, matrix operand order checked by reading, not run on hardware
//
#include <hip/hip_runtime.h>
#include <hip/hip_bf16.h>

// ---------------------------------------------------------------------------
// LinearRTU fused pipeline for gfx1250 (MI455X), wave32 + WMMA bf16.
//   u1 = (x@U1)@V1 ; u2 = (x@U2)@V2  (bf16 WMMA, f32 accumulate)
//   bx = u1*gamma  ; by = u2*gamma
//   c  <- lambda*c + b  rotation scan along T (chunked, analytic carries)
// GEMM: 256-thread block = 8 waves -> 128x64 tile; B panel double-buffered in
// LDS via CDNA5 async-to-LDS copies overlapped with WMMA; K templated so the
// slice loop fully unrolls; B fragments software-pipelined (two live groups);
// __launch_bounds__(256,1) trades occupancy for registers (no acc spills).
// ---------------------------------------------------------------------------

typedef __attribute__((ext_vector_type(16))) __bf16 v16bf;
typedef __attribute__((ext_vector_type(8)))  __bf16 v8bf;
typedef __attribute__((ext_vector_type(4)))  __bf16 v4bf;
typedef __attribute__((ext_vector_type(8)))  float  v8f;

#define B_ 8
#define T_ 2048
#define D_ 1024
#define H_ 1024
#define R_ 128
#define M_ (B_ * T_)        // 16384 rows
#define CL 128              // scan chunk length
#define NC (T_ / CL)        // 16 chunks

#define KS 128              // GEMM K slice staged in LDS
#define LDS_STRIDE 136      // 128 + 8 pad: 272B row stride -> conflict-free b128

// sched_group_barrier masks
#define SGB_WMMA    0x008
#define SGB_VMEMRD  0x020
#define SGB_DSRD    0x100

// ---------------------------------------------------------------------------
// CDNA5 async global->LDS copy (16B per lane), ASYNCcnt tracked. The IOFFSET
// immediate adds to BOTH the LDS and global addresses (ISA 10.x pseudocode).
// ---------------------------------------------------------------------------
template <int OFF>
__device__ __forceinline__ void async_load_b128(unsigned lds_off, const void* gptr) {
    asm volatile("global_load_async_to_lds_b128 %0, %1, off offset:%2"
                 :: "v"(lds_off), "v"(gptr), "n"(OFF)
                 : "memory");
}
__device__ __forceinline__ void wait_async0() {
    asm volatile("s_wait_asynccnt 0x0" ::: "memory");
}

// ---------------------------------------------------------------------------
// Per-channel recurrence parameters: g, phi, gamma.
// ---------------------------------------------------------------------------
__global__ void param_kernel(const float* __restrict__ nu_log,
                             const float* __restrict__ theta_log,
                             float* __restrict__ g, float* __restrict__ phi,
                             float* __restrict__ gamma) {
    int h = blockIdx.x * blockDim.x + threadIdx.x;
    if (h >= H_) return;
    float r  = __expf(-__expf(nu_log[h]));
    float th = __expf(theta_log[h]);
    float s, c;
    __sincosf(th, &s, &c);
    g[h]   = r * c;
    phi[h] = r * s;
    float t = 1.0f - r * r;
    gamma[h] = sqrtf(t > 0.0f ? t : 0.0f);
}

// ---------------------------------------------------------------------------
// f32 -> bf16 conversion of x (4 elements/thread).
// ---------------------------------------------------------------------------
__global__ void cvt_x_kernel(const float* __restrict__ x,
                             __bf16* __restrict__ xb, int n4) {
    int i = blockIdx.x * blockDim.x + threadIdx.x;
    if (i >= n4) return;
    float4 v = ((const float4*)x)[i];
    v4bf o;
    o[0] = (__bf16)v.x; o[1] = (__bf16)v.y;
    o[2] = (__bf16)v.z; o[3] = (__bf16)v.w;
    *(v4bf*)(xb + (size_t)i * 4) = o;
}

// ---------------------------------------------------------------------------
// Transpose + convert weights: src (rows x cols, f32) -> dst (cols x rows, bf16).
// ---------------------------------------------------------------------------
__global__ void cvt_transpose_kernel(const float* __restrict__ src,
                                     __bf16* __restrict__ dst,
                                     int rows, int cols) {
    int i = blockIdx.x * blockDim.x + threadIdx.x;
    if (i >= rows * cols) return;
    int r = i / cols, c = i - r * cols;
    dst[(size_t)c * rows + r] = (__bf16)src[i];
}

union frag_t { v16bf v; v8bf h[2]; };

// ---------------------------------------------------------------------------
// One 128-deep K slice: 4 k-steps x 4 WMMAs, B fragments software-pipelined.
// ---------------------------------------------------------------------------
__device__ __forceinline__ void compute_slice(const __bf16* __restrict__ Aseg,
                                              const __bf16* __restrict__ Bsl,
                                              int klo, int l15, v8f acc[4]) {
    // all 4 A fragments of this slice: one 8-load global clause
    frag_t a[4];
#pragma unroll
    for (int q = 0; q < 4; ++q) {
        a[q].h[0] = *(const v8bf*)(Aseg + q * 32 + klo);
        a[q].h[1] = *(const v8bf*)(Aseg + q * 32 + 16 + klo);
    }
    // prime B group 0
    frag_t b[4];
#pragma unroll
    for (int j = 0; j < 4; ++j) {
        const __bf16* Brow = Bsl + ((j << 4) + l15) * LDS_STRIDE;
        b[j].h[0] = *(const v8bf*)(Brow + klo);
        b[j].h[1] = *(const v8bf*)(Brow + 16 + klo);
    }
#pragma unroll
    for (int q = 0; q < 4; ++q) {
        frag_t bn[4];
        if (q < 3) {                       // prefetch next B group (distinct regs)
#pragma unroll
            for (int j = 0; j < 4; ++j) {
                const __bf16* Brow = Bsl + ((j << 4) + l15) * LDS_STRIDE + (q + 1) * 32;
                bn[j].h[0] = *(const v8bf*)(Brow + klo);
                bn[j].h[1] = *(const v8bf*)(Brow + 16 + klo);
            }
        }
#pragma unroll
        for (int j = 0; j < 4; ++j)
            acc[j] = __builtin_amdgcn_wmma_f32_16x16x32_bf16(
                false, a[q].v, false, b[j].v, (short)0, acc[j], false, false);
        if (q < 3) {
#pragma unroll
            for (int j = 0; j < 4; ++j) b[j] = bn[j];
        }
    }
    // pin the pipeline: A clause, B0, {B(q+1) | WMMA(q)} x3, WMMA(3)
    __builtin_amdgcn_sched_group_barrier(SGB_VMEMRD, 8, 0);
    __builtin_amdgcn_sched_group_barrier(SGB_DSRD,   8, 0);
    __builtin_amdgcn_sched_group_barrier(SGB_DSRD,   8, 0);
    __builtin_amdgcn_sched_group_barrier(SGB_WMMA,   4, 0);
    __builtin_amdgcn_sched_group_barrier(SGB_DSRD,   8, 0);
    __builtin_amdgcn_sched_group_barrier(SGB_WMMA,   4, 0);
    __builtin_amdgcn_sched_group_barrier(SGB_DSRD,   8, 0);
    __builtin_amdgcn_sched_group_barrier(SGB_WMMA,   4, 0);
    __builtin_amdgcn_sched_group_barrier(SGB_WMMA,   4, 0);
}

// ---------------------------------------------------------------------------
// WMMA GEMM, K known at compile time -> fully unrolled slice loop, constant
// LDS buffer selection, immediate addressing throughout.
// grid = (M/128, N/64); block = 256 threads (8 waves, wave w -> 16-row slice).
// MODE 0: store bf16. MODE 1: store f32 scaled by gamma[n].
// __launch_bounds__(256, 1): latency is hidden by the async double buffer and
// WMMA pipeline, not wave count -> give the RA the full register file.
// ---------------------------------------------------------------------------
template <int MODE, int KC>
__global__ void __launch_bounds__(256, 1)
gemm_wmma_kernel(const __bf16* __restrict__ A,
                 const __bf16* __restrict__ Bt,
                 void* __restrict__ out,
                 const float* __restrict__ gamma,
                 int N) {
    __shared__ __bf16 Bs[2][64 * LDS_STRIDE];     // 2 x ~17 KB, bank-padded

    const int tid  = threadIdx.x;
    const int lane = tid & 31;
    const int wv   = tid >> 5;                    // 0..7
    const int l15  = lane & 15;
    const int klo  = (lane >> 4) << 3;            // 0 or 8

    const int mTile = blockIdx.x * 8 + wv;        // 16-row tile per wave
    const int n64   = blockIdx.y << 6;

    const __bf16* Arow = A + (size_t)((mTile << 4) + l15) * KC;

    // staging map: 256 threads x 32 elements cover the 64x128 panel slice
    const int srow = tid >> 2;                    // 0..63
    const int scol = (tid & 3) << 5;              // 0,32,64,96
    const unsigned lds0 = (unsigned)(size_t)(__bf16*)&Bs[0][0];
    const unsigned lds1 = (unsigned)(size_t)(__bf16*)&Bs[1][0];
    const unsigned soff = (unsigned)(srow * LDS_STRIDE + scol) * 2u;
    const __bf16* gbase = Bt + (size_t)(n64 + srow) * KC + scol;

    auto stage = [&](unsigned ldsbase, int kb) {
        const void* g = gbase + kb;
        unsigned l = ldsbase + soff;
        async_load_b128<0>(l, g);
        async_load_b128<16>(l, g);
        async_load_b128<32>(l, g);
        async_load_b128<48>(l, g);
    };

    constexpr int nSlices = KC / KS;

    stage(lds0, 0);                               // prime buffer 0
    wait_async0();
    __syncthreads();

    v8f acc[4] = {};
#pragma unroll
    for (int s = 0; s < nSlices; ++s) {
        const __bf16* BsCur = (s & 1) ? &Bs[1][0] : &Bs[0][0];   // compile-time
        if (s + 1 < nSlices)                      // overlap next copy w/ compute
            stage((s & 1) ? lds0 : lds1, (s + 1) * KS);
        compute_slice(Arow + s * KS, BsCur, klo, l15, acc);
        if (s + 1 < nSlices) {
            wait_async0();                        // next buffer landed
            __syncthreads();
        }
    }

    // C/D layout: lane 0-15 -> M = v, lane 16-31 -> M = 8+v ; N = lane&15.
    const int rowsel = (lane >> 4) << 3;
#pragma unroll
    for (int j = 0; j < 4; ++j) {
        int n = n64 + (j << 4) + l15;
        if (MODE == 0) {
            __bf16* O = (__bf16*)out;
#pragma unroll
            for (int v = 0; v < 8; ++v) {
                int mr = (mTile << 4) + rowsel + v;
                O[(size_t)mr * N + n] = (__bf16)acc[j][v];
            }
        } else {
            float* O = (float*)out;
            float gm = gamma[n];
#pragma unroll
            for (int v = 0; v < 8; ++v) {
                int mr = (mTile << 4) + rowsel + v;
                O[(size_t)mr * N + n] = acc[j][v] * gm;
            }
        }
    }
}

// ---------------------------------------------------------------------------
// Scan phase A: per (b, chunk, h) compute chunk carry sum_s lambda^{L-1-s} b_s.
// ---------------------------------------------------------------------------
__global__ void scan_carry_kernel(const float* __restrict__ bx,
                                  const float* __restrict__ by,
                                  const float* __restrict__ g,
                                  const float* __restrict__ phi,
                                  float2* __restrict__ carry /* [B][NC][H] */) {
    int idx = blockIdx.x * blockDim.x + threadIdx.x;
    if (idx >= B_ * NC * H_) return;
    int h = idx % H_;
    int c = (idx / H_) % NC;
    int b = idx / (H_ * NC);
    float lg = g[h], lp = phi[h];
    const float* px = bx + ((size_t)b * T_ + (size_t)c * CL) * H_ + h;
    const float* py = by + ((size_t)b * T_ + (size_t)c * CL) * H_ + h;
    float ax = 0.0f, ay = 0.0f;
    for (int t = 0; t < CL; ++t) {
        float vx = px[(size_t)t * H_];
        float vy = py[(size_t)t * H_];
        float nx = lg * ax - lp * ay + vx;
        float ny = lp * ax + lg * ay + vy;
        ax = nx; ay = ny;
    }
    carry[((size_t)b * NC + c) * H_ + h] = make_float2(ax, ay);
}

// ---------------------------------------------------------------------------
// Scan phase B: per (b,h) combine NC chunk carries with analytic lambda^CL;
// emits each chunk's incoming state and the final states.
// ---------------------------------------------------------------------------
__global__ void scan_state_kernel(const float* __restrict__ nu_log,
                                  const float* __restrict__ theta_log,
                                  const float* __restrict__ hc1,
                                  const float* __restrict__ hc2,
                                  const float2* __restrict__ carry,
                                  float2* __restrict__ states, /* [B][NC][H] */
                                  float* __restrict__ out_tail) {
    int idx = blockIdx.x * blockDim.x + threadIdx.x;
    if (idx >= B_ * H_) return;
    int h = idx % H_;
    int b = idx / H_;
    float nu = __expf(nu_log[h]);
    float th = __expf(theta_log[h]);
    float rL = __expf(-(float)CL * nu);
    float s, cc;
    __sincosf((float)CL * th, &s, &cc);
    float gL = rL * cc, pL = rL * s;
    float sx = hc1[idx], sy = hc2[idx];
    for (int c = 0; c < NC; ++c) {
        states[((size_t)b * NC + c) * H_ + h] = make_float2(sx, sy);
        float2 cr = carry[((size_t)b * NC + c) * H_ + h];
        float nx = gL * sx - pL * sy + cr.x;
        float ny = pL * sx + gL * sy + cr.y;
        sx = nx; sy = ny;
    }
    out_tail[(size_t)b * H_ + h] = sx;                       // c1[:, -1, :]
    out_tail[(size_t)B_ * H_ + (size_t)b * H_ + h] = sy;     // c2[:, -1, :]
}

// ---------------------------------------------------------------------------
// Scan phase C: re-run each chunk seeded with its incoming state; single write
// of y (no RMW on the 128 MiB output).
// ---------------------------------------------------------------------------
__global__ void scan_apply_kernel(const float* __restrict__ bx,
                                  const float* __restrict__ by,
                                  const float* __restrict__ g,
                                  const float* __restrict__ phi,
                                  const float2* __restrict__ states,
                                  float* __restrict__ y /* [B][T][2H] */) {
    int idx = blockIdx.x * blockDim.x + threadIdx.x;
    if (idx >= B_ * NC * H_) return;
    int h = idx % H_;
    int c = (idx / H_) % NC;
    int b = idx / (H_ * NC);
    float lg = g[h], lp = phi[h];
    float2 S = states[((size_t)b * NC + c) * H_ + h];
    float ax = S.x, ay = S.y;
    const float* px = bx + ((size_t)b * T_ + (size_t)c * CL) * H_ + h;
    const float* py = by + ((size_t)b * T_ + (size_t)c * CL) * H_ + h;
    float* yo = y + ((size_t)b * T_ + (size_t)c * CL) * (2 * H_);
    for (int t = 0; t < CL; ++t) {
        float nx = lg * ax - lp * ay + px[(size_t)t * H_];
        float ny = lp * ax + lg * ay + py[(size_t)t * H_];
        ax = nx; ay = ny;
        yo[(size_t)t * (2 * H_) + h]      = ax;
        yo[(size_t)t * (2 * H_) + H_ + h] = ay;
    }
}

// ---------------------------------------------------------------------------
// Host-side orchestration (all on `stream`, graph-capture safe).
// ---------------------------------------------------------------------------
extern "C" void kernel_launch(void* const* d_in, const int* in_sizes, int n_in,
                              void* d_out, int out_size, void* d_ws, size_t ws_size,
                              hipStream_t stream) {
    const float* x         = (const float*)d_in[0];
    const float* nu_log    = (const float*)d_in[1];
    const float* theta_log = (const float*)d_in[2];
    const float* U1        = (const float*)d_in[3];
    const float* U2        = (const float*)d_in[4];
    const float* V1        = (const float*)d_in[5];
    const float* V2        = (const float*)d_in[6];
    const float* hc1       = (const float*)d_in[7];
    const float* hc2       = (const float*)d_in[8];

    // ---- workspace carve-up (256B aligned slices) ----
    char* ws = (char*)d_ws;
    size_t off = 0;
    auto alloc = [&](size_t bytes) {
        char* p = ws + off;
        off += (bytes + 255) & ~(size_t)255;
        return p;
    };
    float*  g_p     = (float*)alloc(H_ * sizeof(float));
    float*  phi_p   = (float*)alloc(H_ * sizeof(float));
    float*  gam_p   = (float*)alloc(H_ * sizeof(float));
    __bf16* xb      = (__bf16*)alloc((size_t)M_ * D_ * sizeof(__bf16));   // 32 MiB
    __bf16* U1t     = (__bf16*)alloc((size_t)R_ * D_ * sizeof(__bf16));
    __bf16* U2t     = (__bf16*)alloc((size_t)R_ * D_ * sizeof(__bf16));
    __bf16* V1t     = (__bf16*)alloc((size_t)H_ * R_ * sizeof(__bf16));
    __bf16* V2t     = (__bf16*)alloc((size_t)H_ * R_ * sizeof(__bf16));
    __bf16* t1      = (__bf16*)alloc((size_t)M_ * R_ * sizeof(__bf16));   // 4 MiB
    __bf16* t2      = (__bf16*)alloc((size_t)M_ * R_ * sizeof(__bf16));
    float*  bx      = (float*)alloc((size_t)M_ * H_ * sizeof(float));     // 64 MiB
    float*  by      = (float*)alloc((size_t)M_ * H_ * sizeof(float));     // 64 MiB
    float2* carry   = (float2*)alloc((size_t)B_ * NC * H_ * sizeof(float2));
    float2* states  = (float2*)alloc((size_t)B_ * NC * H_ * sizeof(float2));

    float* y    = (float*)d_out;                       // [B][T][2H]
    float* tail = y + (size_t)B_ * T_ * 2 * H_;        // c1_last, c2_last

    // 1) per-channel params
    param_kernel<<<(H_ + 255) / 256, 256, 0, stream>>>(nu_log, theta_log,
                                                       g_p, phi_p, gam_p);

    // 2) conversions
    {
        int n4 = (M_ * D_) / 4;
        cvt_x_kernel<<<(n4 + 255) / 256, 256, 0, stream>>>(x, xb, n4);
        int nu_el = D_ * R_;
        cvt_transpose_kernel<<<(nu_el + 255) / 256, 256, 0, stream>>>(U1, U1t, D_, R_);
        cvt_transpose_kernel<<<(nu_el + 255) / 256, 256, 0, stream>>>(U2, U2t, D_, R_);
        int nv_el = R_ * H_;
        cvt_transpose_kernel<<<(nv_el + 255) / 256, 256, 0, stream>>>(V1, V1t, R_, H_);
        cvt_transpose_kernel<<<(nv_el + 255) / 256, 256, 0, stream>>>(V2, V2t, R_, H_);
    }

    // 3) stage-1 GEMMs: t = x @ U  (M=16384, N=128, K=1024) -> bf16
    {
        dim3 grid(M_ / 128, R_ / 64);     // (128, 2)
        gemm_wmma_kernel<0, D_><<<grid, 256, 0, stream>>>(xb, U1t, t1, nullptr, R_);
        gemm_wmma_kernel<0, D_><<<grid, 256, 0, stream>>>(xb, U2t, t2, nullptr, R_);
    }

    // 4) stage-2 GEMMs: b = (t @ V) * gamma  (M=16384, N=1024, K=128) -> f32
    {
        dim3 grid(M_ / 128, H_ / 64);     // (128, 16)
        gemm_wmma_kernel<1, R_><<<grid, 256, 0, stream>>>(t1, V1t, bx, gam_p, H_);
        gemm_wmma_kernel<1, R_><<<grid, 256, 0, stream>>>(t2, V2t, by, gam_p, H_);
    }

    // 5) chunked rotation scan
    {
        int nA = B_ * NC * H_;                  // 131072 threads
        scan_carry_kernel<<<(nA + 255) / 256, 256, 0, stream>>>(bx, by, g_p, phi_p, carry);
        int nB = B_ * H_;                       // 8192 threads
        scan_state_kernel<<<(nB + 255) / 256, 256, 0, stream>>>(nu_log, theta_log,
                                                                hc1, hc2, carry,
                                                                states, tail);
        scan_apply_kernel<<<(nA + 255) / 256, 256, 0, stream>>>(bx, by, g_p, phi_p,
                                                                states, y);
    }
    (void)in_sizes; (void)n_in; (void)out_size; (void)ws_size;
}
